// ResidualBiGRU_57793079935580
// MI455X (gfx1250) — compile-verified, matrix operand order checked by reading, not run on hardware
//
#include <hip/hip_runtime.h>
#include <math.h>

#define BB 32
#define TT 2048
#define MMc 64
#define DDc 256
#define HHc 256

typedef __attribute__((ext_vector_type(16))) __bf16 v16bf;
typedef __attribute__((ext_vector_type(8)))  float  v8f;

#if defined(__gfx1250__) && __has_builtin(__builtin_amdgcn_tensor_load_to_lds)
#define USE_TDM 1
#endif

__device__ __forceinline__ unsigned short f2bf(float f) {
  unsigned int u = __float_as_uint(f);
  u += 0x7FFFu + ((u >> 16) & 1u);
  return (unsigned short)(u >> 16);
}
__device__ __forceinline__ float sigmoidf_(float x) { return 1.0f / (1.0f + expf(-x)); }
__device__ __forceinline__ float gelu_(float x) { return 0.5f * x * (1.0f + erff(x * 0.70710678118654752f)); }

#ifdef USE_TDM
typedef __attribute__((ext_vector_type(4))) unsigned int t_v4u;
typedef __attribute__((ext_vector_type(8))) int t_v8i;
typedef __attribute__((ext_vector_type(4))) int t_v4i;

// DMA a 128-row x 32-col bf16 tile (row stride ldk elems) from global to LDS.
// D# group0: count=1, lds_addr, global_addr, type=2("image")
// D# group1: data_size=1(2B), tensor_dim0=ldk, tensor_dim1=128, tile_dim0=32,
//            tile_dim1=128, tensor_dim0_stride=ldk
__device__ __forceinline__ void tdm_load2d(const unsigned short* gsrc, unsigned lds_off,
                                           int ldk) {
  unsigned long long ga = (unsigned long long)(size_t)gsrc;
  t_v4u g0 = { 1u, lds_off, (unsigned)ga,
               (unsigned)(((ga >> 32) & 0x01FFFFFFull) | 0x80000000u) };
  t_v8i g1 = { (int)(1u << 16),
               (int)(((unsigned)ldk & 0xFFFFu) << 16),
               (int)((((unsigned)ldk >> 16) & 0xFFFFu) | (128u << 16)),
               (int)(32u << 16),
               (int)128,
               (int)ldk,
               0, 0 };
  t_v4i gz = { 0, 0, 0, 0 };
#if __clang_major__ >= 23
  t_v8i gz8 = { 0, 0, 0, 0, 0, 0, 0, 0 };
  __builtin_amdgcn_tensor_load_to_lds(g0, g1, gz, gz, gz8, 0);
#else
  __builtin_amdgcn_tensor_load_to_lds(g0, g1, gz, gz, 0);
#endif
}
#endif

// ---------------- elementwise cast f32 -> bf16 ----------------
__global__ __launch_bounds__(256) void k_castw(const float* __restrict__ in,
                                               unsigned short* __restrict__ out, int n) {
  int i = blockIdx.x * 256 + threadIdx.x;
  if (i < n) out[i] = f2bf(in[i]);
}

// ---------------- concat [y|xp] -> bf16 X0 (B*T,320) ----------------
__global__ __launch_bounds__(256) void k_concat(const float* __restrict__ y,
                                                const float* __restrict__ xp,
                                                unsigned short* __restrict__ X0) {
  int idx = blockIdx.x * 256 + threadIdx.x;           // < 65536*320
  int rt = idx / 320;
  int c  = idx - rt * 320;
  float v = (c < MMc) ? y[(size_t)rt * MMc + c] : xp[(size_t)rt * DDc + (c - MMc)];
  X0[idx] = f2bf(v);
}

// ---------------- GEMM: C(R,N) = A(R,K)bf16 @ W(N,K)bf16^T + bias, opt GELU -------------
// block: 256 thr (8 waves), tile 128(M) x 128(N), K step 32, v_wmma_f32_16x16x32_bf16.
// Tiles staged via TDM tensor_load_to_lds (double-buffered, TENSORcnt) when available.
__global__ __launch_bounds__(256) void k_gemm(const unsigned short* __restrict__ A,
                                              const unsigned short* __restrict__ W,
                                              const float* __restrict__ bias,
                                              float* __restrict__ outf,
                                              unsigned short* __restrict__ outb,
                                              int R, int K, int N, int act) {
  __shared__ unsigned short At[2][128 * 32];
  __shared__ unsigned short Bt[2][128 * 32];
  const int r0 = blockIdx.x * 128;
  const int n0 = blockIdx.y * 128;
  const int tid = threadIdx.x;
  const int wave = tid >> 5, lane = tid & 31;
  const int wm = wave & 3, wn = wave >> 2;            // 4x2 wave grid of 32x64 sub-tiles
  const int lrow = lane & 15, koff = (lane >> 4) * 16;
  const int nk = K >> 5;

  v8f acc[2][4];
#pragma unroll
  for (int i = 0; i < 2; i++)
#pragma unroll
    for (int j = 0; j < 4; j++) acc[i][j] = {};

#ifdef USE_TDM
  if (wave == 0) {
    tdm_load2d(A + (size_t)r0 * K, (unsigned)(size_t)&At[0][0], K);
    tdm_load2d(W + (size_t)n0 * K, (unsigned)(size_t)&Bt[0][0], K);
  }
#else
  {
    int row = tid >> 1, half = tid & 1;
    *(uint4*)(&At[0][row * 32 + half * 16]) =
        *(const uint4*)(A + (size_t)(r0 + row) * K + half * 16);
    *(uint4*)(&Bt[0][row * 32 + half * 16]) =
        *(const uint4*)(W + (size_t)(n0 + row) * K + half * 16);
  }
#endif

  for (int kk = 0; kk < nk; kk++) {
    const int cur = kk & 1, nxt = cur ^ 1;
#ifdef USE_TDM
    if (wave == 0) {
      if (kk + 1 < nk) {
        tdm_load2d(A + (size_t)r0 * K + (kk + 1) * 32, (unsigned)(size_t)&At[nxt][0], K);
        tdm_load2d(W + (size_t)n0 * K + (kk + 1) * 32, (unsigned)(size_t)&Bt[nxt][0], K);
        __builtin_amdgcn_s_wait_tensorcnt(2);   // current pair complete
      } else {
        __builtin_amdgcn_s_wait_tensorcnt(0);
      }
    }
    __syncthreads();
#else
    __syncthreads();
    if (kk + 1 < nk) {
      int k0 = (kk + 1) * 32;
      int row = tid >> 1, half = tid & 1;
      *(uint4*)(&At[nxt][row * 32 + half * 16]) =
          *(const uint4*)(A + (size_t)(r0 + row) * K + k0 + half * 16);
      *(uint4*)(&Bt[nxt][row * 32 + half * 16]) =
          *(const uint4*)(W + (size_t)(n0 + row) * K + k0 + half * 16);
    }
#endif
    v16bf af[2], bfr[4];
#pragma unroll
    for (int i = 0; i < 2; i++)
      af[i] = *(const v16bf*)(&At[cur][(wm * 32 + i * 16 + lrow) * 32 + koff]);
#pragma unroll
    for (int j = 0; j < 4; j++)
      bfr[j] = *(const v16bf*)(&Bt[cur][(wn * 64 + j * 16 + lrow) * 32 + koff]);
#pragma unroll
    for (int i = 0; i < 2; i++)
#pragma unroll
      for (int j = 0; j < 4; j++)
        acc[i][j] = __builtin_amdgcn_wmma_f32_16x16x32_bf16(
            false, af[i], false, bfr[j], (short)0, acc[i][j], false, false);
    __syncthreads();
  }

  const int hi = lane >> 4;
#pragma unroll
  for (int i = 0; i < 2; i++)
#pragma unroll
    for (int j = 0; j < 4; j++) {
      int col = n0 + wn * 64 + j * 16 + lrow;
      float bv = bias ? bias[col] : 0.0f;
#pragma unroll
      for (int e = 0; e < 8; e++) {
        int row = r0 + wm * 32 + i * 16 + e + 8 * hi;
        float v = acc[i][j][e] + bv;
        if (act == 1) v = gelu_(v);
        size_t o = (size_t)row * N + col;
        if (outf) outf[o] = v;
        if (outb) outb[o] = f2bf(v);
      }
    }
  (void)R;
}

// ---------------- GRU scan: 32 WGs = 2 dirs x 16 hidden-slices of 16 units ----------------
// block: 192 thr (6 waves). Whh slice (48x256 bf16) resident in LDS. Per step:
// WMMA 32x48 GEMM, gate math, publish h slice via L2, atomic-counter barrier.
__global__ __launch_bounds__(192) void k_scan(const unsigned short* __restrict__ Whh_f,
                                              const unsigned short* __restrict__ Whh_b,
                                              const float* __restrict__ bhh_f,
                                              const float* __restrict__ bhh_b,
                                              const float* __restrict__ gx_f,
                                              const float* __restrict__ gx_b,
                                              float* __restrict__ hs,
                                              unsigned short* __restrict__ hsb,
                                              float* __restrict__ hstate,
                                              int* __restrict__ syncc) {
  const int g = blockIdx.x;
  const int dir = g >> 4;
  const int sl = g & 15;
  const int j0 = sl * 16;
  const int tid = threadIdx.x;

  __shared__ unsigned short wlds[48 * 256];   // 24 KB  rows = [r|z|n] x 16 units
  __shared__ unsigned short hb[32 * 256];     // 16 KB  full h, bf16 (A operand)
  __shared__ float ghl[32 * 48];              //  6 KB  gh slice, f32
  __shared__ float hf[32 * 16];               //  2 KB  local h slice, f32
  __shared__ float bh[48];

  const unsigned short* Whh = dir ? Whh_b : Whh_f;
  const float* bhh = dir ? bhh_b : bhh_f;
  const float* gx  = dir ? gx_b  : gx_f;

  // one-time: load resident weight slice (rows gate*256 + j0 + jj)
  for (int c = tid; c < 1536; c += 192) {     // 16B chunks
    int lr = c >> 5, cc = (c & 31) * 8;
    int gate = lr >> 4, jj = lr & 15;
    *(uint4*)(wlds + lr * 256 + cc) =
        *(const uint4*)(Whh + (size_t)(gate * HHc + j0 + jj) * HHc + cc);
  }
  if (tid < 48) { int gate = tid >> 4, jj = tid & 15; bh[tid] = bhh[gate * HHc + j0 + jj]; }
  for (int c = tid; c < 32 * 256; c += 192) hb[c] = 0;
  for (int c = tid; c < 32 * 16; c += 192) hf[c] = 0.0f;
  __syncthreads();

  const int wave = tid >> 5, lane = tid & 31;
  const int mt = wave & 1, nt = wave >> 1;    // 2(M) x 3(N) tiles over 6 waves
  const int lrow = lane & 15, koff = (lane >> 4) * 16;
  const int hi8 = (lane >> 4) * 8;
  float* hst = hstate + (size_t)dir * 8192;
  volatile int* vs = (volatile int*)syncc;

  for (int i = 0; i < TT; i++) {
    const int t = dir ? (TT - 1 - i) : i;

    v8f acc = {};
#pragma unroll
    for (int kc = 0; kc < 8; kc++) {
      v16bf a = *(const v16bf*)(hb   + (mt * 16 + lrow) * 256 + kc * 32 + koff);
      v16bf b = *(const v16bf*)(wlds + (nt * 16 + lrow) * 256 + kc * 32 + koff);
      acc = __builtin_amdgcn_wmma_f32_16x16x32_bf16(false, a, false, b, (short)0, acc,
                                                    false, false);
    }
    {
      int col = nt * 16 + lrow;
#pragma unroll
      for (int e = 0; e < 8; e++) ghl[(mt * 16 + e + hi8) * 48 + col] = acc[e];
    }
    __syncthreads();

    // gate math for 32 batches x 16 units
    for (int idx = tid; idx < 512; idx += 192) {
      int b_ = idx >> 4, jj = idx & 15;
      int j = j0 + jj;
      float ghr = ghl[b_ * 48 + jj]      + bh[jj];
      float ghz = ghl[b_ * 48 + 16 + jj] + bh[16 + jj];
      float ghn = ghl[b_ * 48 + 32 + jj] + bh[32 + jj];
      size_t gb = ((size_t)b_ * TT + t) * 768;
      float r = sigmoidf_(gx[gb + j]           + ghr);
      float z = sigmoidf_(gx[gb + HHc + j]     + ghz);
      float n = tanhf   (gx[gb + 2 * HHc + j]  + r * ghn);
      float hp = hf[b_ * 16 + jj];
      float hnw = (1.0f - z) * n + z * hp;
      hf[b_ * 16 + jj] = hnw;
      size_t o = ((size_t)b_ * TT + t) * 512 + dir * HHc + j;
      hs[o] = hnw;
      hsb[o] = f2bf(hnw);
      hst[b_ * HHc + j] = hnw;
    }
    __threadfence();
    __syncthreads();
    if (tid == 0) {
      atomicAdd(&syncc[dir], 1);
      int target = 16 * (i + 1);
      while (vs[dir] < target) __builtin_amdgcn_s_sleep(2);
    }
    __syncthreads();
    __threadfence();

    // restage full h as bf16 for next step's WMMA A operand
    for (int c = tid; c < 2048; c += 192) {
      float4 v = *(const float4*)(hst + c * 4);
      uint2 p;
      p.x = (unsigned int)f2bf(v.x) | ((unsigned int)f2bf(v.y) << 16);
      p.y = (unsigned int)f2bf(v.z) | ((unsigned int)f2bf(v.w) << 16);
      *(uint2*)(hb + c * 4) = p;
    }
    __syncthreads();
  }
}

// ---------------- LayerNorm over 512, f32 in -> bf16 out ----------------
__global__ __launch_bounds__(256) void k_ln(const float* __restrict__ hsin,
                                            const float* __restrict__ gw,
                                            const float* __restrict__ bw,
                                            unsigned short* __restrict__ hn) {
  __shared__ float red[256];
  int row = blockIdx.x, tid = threadIdx.x;
  const float* x = hsin + (size_t)row * 512;
  float a0 = x[tid], a1 = x[tid + 256];
  red[tid] = a0 + a1;
  __syncthreads();
  for (int s = 128; s > 0; s >>= 1) { if (tid < s) red[tid] += red[tid + s]; __syncthreads(); }
  float mu = red[0] * (1.0f / 512.0f);
  __syncthreads();
  float d0 = a0 - mu, d1 = a1 - mu;
  red[tid] = d0 * d0 + d1 * d1;
  __syncthreads();
  for (int s = 128; s > 0; s >>= 1) { if (tid < s) red[tid] += red[tid + s]; __syncthreads(); }
  float rs = rsqrtf(red[0] * (1.0f / 512.0f) + 1e-5f);
  unsigned short* o = hn + (size_t)row * 512;
  o[tid]       = f2bf(d0 * rs * gw[tid]       + bw[tid]);
  o[tid + 256] = f2bf(d1 * rs * gw[tid + 256] + bw[tid + 256]);
}

// ------------------------------ host ------------------------------
extern "C" void kernel_launch(void* const* d_in, const int* in_sizes, int n_in,
                              void* d_out, int out_size, void* d_ws, size_t ws_size,
                              hipStream_t stream) {
  (void)in_sizes; (void)n_in; (void)out_size; (void)ws_size;
  const float* y     = (const float*)d_in[0];
  const float* xp    = (const float*)d_in[1];
  const float* Wih0f = (const float*)d_in[2];  const float* Whh0f = (const float*)d_in[3];
  const float* bih0f = (const float*)d_in[4];  const float* bhh0f = (const float*)d_in[5];
  const float* Wih0b = (const float*)d_in[6];  const float* Whh0b = (const float*)d_in[7];
  const float* bih0b = (const float*)d_in[8];  const float* bhh0b = (const float*)d_in[9];
  const float* Wih1f = (const float*)d_in[10]; const float* Whh1f = (const float*)d_in[11];
  const float* bih1f = (const float*)d_in[12]; const float* bhh1f = (const float*)d_in[13];
  const float* Wih1b = (const float*)d_in[14]; const float* Whh1b = (const float*)d_in[15];
  const float* bih1b = (const float*)d_in[16]; const float* bhh1b = (const float*)d_in[17];
  const float* ln_g  = (const float*)d_in[18]; const float* ln_b  = (const float*)d_in[19];
  const float* W1    = (const float*)d_in[20]; const float* b1    = (const float*)d_in[21];
  const float* W2    = (const float*)d_in[22]; const float* b2    = (const float*)d_in[23];
  float* out = (float*)d_out;

  char* ws = (char*)d_ws;
  size_t off = 0;
  auto alloc = [&](size_t bytes) { size_t r = off; off = (off + bytes + 255) & ~(size_t)255; return r; };
  const size_t RT = (size_t)BB * TT;          // 65536 rows

  size_t o_wih0f = alloc(768 * 320 * 2), o_wih0b = alloc(768 * 320 * 2);
  size_t o_whh0f = alloc(768 * 256 * 2), o_whh0b = alloc(768 * 256 * 2);
  size_t o_wih1f = alloc(768 * 512 * 2), o_wih1b = alloc(768 * 512 * 2);
  size_t o_whh1f = alloc(768 * 256 * 2), o_whh1b = alloc(768 * 256 * 2);
  size_t o_w1    = alloc(512 * 512 * 2), o_w2    = alloc(256 * 512 * 2);
  size_t o_x0    = alloc(RT * 320 * 2);
  size_t o_gxf   = alloc(RT * 768 * 4);
  size_t o_gxb   = alloc(RT * 768 * 4);
  size_t o_hsf   = alloc(RT * 512 * 4);       // layer0 f32 out, reused by layer1
  size_t o_hsb   = alloc(RT * 512 * 2);       // bf16 out (layer1's input)
  size_t o_hn    = alloc(RT * 512 * 2);
  size_t o_zb    = alloc(RT * 512 * 2);
  size_t o_hst   = alloc(2 * 8192 * 4 + 256); // hstate + sync counters
  float* hstate = (float*)(ws + o_hst);
  int*   syncc  = (int*)(ws + o_hst + 2 * 8192 * 4);

  auto castw = [&](const float* src, size_t dst, int n) {
    k_castw<<<(n + 255) / 256, 256, 0, stream>>>(src, (unsigned short*)(ws + dst), n);
  };
  castw(Wih0f, o_wih0f, 768 * 320); castw(Wih0b, o_wih0b, 768 * 320);
  castw(Whh0f, o_whh0f, 768 * 256); castw(Whh0b, o_whh0b, 768 * 256);
  castw(Wih1f, o_wih1f, 768 * 512); castw(Wih1b, o_wih1b, 768 * 512);
  castw(Whh1f, o_whh1f, 768 * 256); castw(Whh1b, o_whh1b, 768 * 256);
  castw(W1, o_w1, 512 * 512);       castw(W2, o_w2, 256 * 512);

  k_concat<<<(int)(RT * 320 / 256), 256, 0, stream>>>(y, xp, (unsigned short*)(ws + o_x0));

  // layer0 gx (R=65536, K=320, N=768)
  k_gemm<<<dim3(512, 6), 256, 0, stream>>>((unsigned short*)(ws + o_x0),
      (unsigned short*)(ws + o_wih0f), bih0f, (float*)(ws + o_gxf), nullptr,
      (int)RT, 320, 768, 0);
  k_gemm<<<dim3(512, 6), 256, 0, stream>>>((unsigned short*)(ws + o_x0),
      (unsigned short*)(ws + o_wih0b), bih0b, (float*)(ws + o_gxb), nullptr,
      (int)RT, 320, 768, 0);

  hipMemsetAsync(ws + o_hst, 0, 2 * 8192 * 4 + 256, stream);
  k_scan<<<32, 192, 0, stream>>>((unsigned short*)(ws + o_whh0f), (unsigned short*)(ws + o_whh0b),
      bhh0f, bhh0b, (float*)(ws + o_gxf), (float*)(ws + o_gxb),
      (float*)(ws + o_hsf), (unsigned short*)(ws + o_hsb), hstate, syncc);

  // layer1 gx (K=512), reuse gx buffers
  k_gemm<<<dim3(512, 6), 256, 0, stream>>>((unsigned short*)(ws + o_hsb),
      (unsigned short*)(ws + o_wih1f), bih1f, (float*)(ws + o_gxf), nullptr,
      (int)RT, 512, 768, 0);
  k_gemm<<<dim3(512, 6), 256, 0, stream>>>((unsigned short*)(ws + o_hsb),
      (unsigned short*)(ws + o_wih1b), bih1b, (float*)(ws + o_gxb), nullptr,
      (int)RT, 512, 768, 0);

  hipMemsetAsync(ws + o_hst, 0, 2 * 8192 * 4 + 256, stream);
  k_scan<<<32, 192, 0, stream>>>((unsigned short*)(ws + o_whh1f), (unsigned short*)(ws + o_whh1b),
      bhh1f, bhh1b, (float*)(ws + o_gxf), (float*)(ws + o_gxb),
      (float*)(ws + o_hsf), (unsigned short*)(ws + o_hsb), hstate, syncc);

  k_ln<<<(int)RT, 256, 0, stream>>>((float*)(ws + o_hsf), ln_g, ln_b,
                                    (unsigned short*)(ws + o_hn));

  // MLP: z = GELU(hn @ W1^T + b1) (bf16 out), out = z @ W2^T + b2 (f32)
  k_gemm<<<dim3(512, 4), 256, 0, stream>>>((unsigned short*)(ws + o_hn),
      (unsigned short*)(ws + o_w1), b1, nullptr, (unsigned short*)(ws + o_zb),
      (int)RT, 512, 512, 1);
  k_gemm<<<dim3(512, 2), 256, 0, stream>>>((unsigned short*)(ws + o_zb),
      (unsigned short*)(ws + o_w2), b2, out, nullptr,
      (int)RT, 512, 256, 0);
}
